// Attention_gate_46213848105841
// MI455X (gfx1250) — matrix operand order repaired and checked by reading, not compiled, not run on hardware
//
#include <hip/hip_runtime.h>
#include <stdint.h>

typedef __bf16 bf16;
typedef __attribute__((ext_vector_type(16))) __bf16 v16bf;
typedef __attribute__((ext_vector_type(8)))  float  v8f;
typedef __attribute__((ext_vector_type(4)))  float  f32x4;   // native vector: OK for nontemporal builtins

#define CIN  256
#define FCH  128
#define NROW (8 * 128 * 128)   // 131072 pixels
#define EPSV 1e-3f

// ---------------- workspace layout (bytes) ----------------
// [0      , 65536)  : Bg fragments, 64 frags (k*8+j) x 1KB, WMMA B layout, bf16
// [65536  , 131072) : Bx fragments, same layout
// [131072 + 0   )   : bias_g[128]  f32  (bg*inv + beta - mean*inv)
// [131072 + 512 )   : bias_x[128]  f32
// [131072 + 1024)   : wp[128]      f32  (Wp * c0)
// [131072 + 1536)   : bp_fold      f32  (bp*c0 + beta_p - mean_p*c0)

__global__ __launch_bounds__(256) void prep_weights(
    const float* __restrict__ Wg, const float* __restrict__ gamma_g, const float* __restrict__ var_g,
    const float* __restrict__ Wx, const float* __restrict__ gamma_x, const float* __restrict__ var_x,
    uint32_t* __restrict__ ws32)
{
  int t = blockIdx.x * 256 + threadIdx.x;          // 32768 threads, one packed dword each
  int mat  = t >> 14;                              // 0 = g, 1 = x
  int rem  = t & 16383;
  int frag = rem >> 8;                             // 0..63  (= k*8 + j)
  int lane = (rem >> 3) & 31;
  int v    = rem & 7;
  int k = frag >> 3, j = frag & 7;
  int n  = j * 16 + (lane & 15);                   // output channel (B column)
  int K0 = k * 32 + ((lane >> 4) << 3) + 2 * (v & 3) + ((v >> 2) << 4);    // B layout: lanes 0-15 K={0..7,16..23}, 16-31 K={8..15,24..31}

  const float* W  = mat ? Wx : Wg;
  const float* ga = mat ? gamma_x : gamma_g;
  const float* va = mat ? var_x : var_g;
  float inv = ga[n] * rsqrtf(va[n] + EPSV);
  float w0 = W[(size_t)K0 * FCH + n] * inv;
  float w1 = W[(size_t)(K0 + 1) * FCH + n] * inv;
  union { __bf16 b[2]; uint32_t u; } p;
  p.b[0] = (__bf16)w0; p.b[1] = (__bf16)w1;
  ws32[t] = p.u;
}

__global__ void prep_vectors(
    const float* __restrict__ bg, const float* __restrict__ gamma_g, const float* __restrict__ beta_g,
    const float* __restrict__ mean_g, const float* __restrict__ var_g,
    const float* __restrict__ bx, const float* __restrict__ gamma_x, const float* __restrict__ beta_x,
    const float* __restrict__ mean_x, const float* __restrict__ var_x,
    const float* __restrict__ Wp, const float* __restrict__ bp,
    const float* __restrict__ gamma_p, const float* __restrict__ beta_p,
    const float* __restrict__ mean_p, const float* __restrict__ var_p,
    float* __restrict__ vws)   // points at ws + 131072
{
  int t = threadIdx.x;
  if (t < FCH) {
    float ig = gamma_g[t] * rsqrtf(var_g[t] + EPSV);
    vws[t] = bg[t] * ig + beta_g[t] - mean_g[t] * ig;
    float ix = gamma_x[t] * rsqrtf(var_x[t] + EPSV);
    vws[128 + t] = bx[t] * ix + beta_x[t] - mean_x[t] * ix;
    float c0 = gamma_p[0] * rsqrtf(var_p[0] + EPSV);
    vws[256 + t] = Wp[t] * c0;
    if (t == 0) vws[384] = bp[0] * c0 + beta_p[0] - mean_p[0] * c0;
  }
}

// Raw fp32 A-tile fetch (per-lane 2x32B, non-temporal: each line touched once).
__device__ inline void load_a_raw(const float* __restrict__ p, f32x4 r[4])
{
  r[0] = __builtin_nontemporal_load((const f32x4*)(p));
  r[1] = __builtin_nontemporal_load((const f32x4*)(p + 4));
  r[2] = __builtin_nontemporal_load((const f32x4*)(p + 16));
  r[3] = __builtin_nontemporal_load((const f32x4*)(p + 20));
}

// Pack raw fp32 into the ISA 16x32 bf16 A-fragment layout.
// lanes 0-15: row=lane, K = {0..7, 16..23}; lanes 16-31: row=lane-16, K = {8..15, 24..31}.
__device__ inline v16bf cvt_a_frag(const f32x4 r[4])
{
  v16bf o;
  o[0]  = (__bf16)r[0].x; o[1]  = (__bf16)r[0].y; o[2]  = (__bf16)r[0].z; o[3]  = (__bf16)r[0].w;
  o[4]  = (__bf16)r[1].x; o[5]  = (__bf16)r[1].y; o[6]  = (__bf16)r[1].z; o[7]  = (__bf16)r[1].w;
  o[8]  = (__bf16)r[2].x; o[9]  = (__bf16)r[2].y; o[10] = (__bf16)r[2].z; o[11] = (__bf16)r[2].w;
  o[12] = (__bf16)r[3].x; o[13] = (__bf16)r[3].y; o[14] = (__bf16)r[3].z; o[15] = (__bf16)r[3].w;
  return o;
}

__global__ __launch_bounds__(256) void attn_gate_main(
    const float* __restrict__ Ag, const float* __restrict__ Ax,
    const char* __restrict__ ws, float* __restrict__ out)
{
  // Stage both folded-weight fragment images in LDS (MI455X: 320KB per WGP).
  __shared__ __align__(32) char Bsm[131072];
  {
    const f32x4* __restrict__ src = (const f32x4*)ws;
    f32x4* dst = (f32x4*)Bsm;
    int t = threadIdx.x;
#pragma unroll
    for (int i = 0; i < 32; ++i)          // 256 threads x 16B, coalesced 4KB/iter
      dst[i * 256 + t] = src[i * 256 + t];
  }
  __syncthreads();

  const int lane = threadIdx.x & 31;
  const int wave = threadIdx.x >> 5;
  const int rowbase = blockIdx.x * 128 + wave * 16;   // 16 rows per wave
  const int arow = rowbase + (lane & 15);
  const float* agp = Ag + (size_t)arow * CIN;
  const float* axp = Ax + (size_t)arow * CIN;
  const int koff = (lane >> 4) * 8;                   // A-fragment half select
  // Two LDS bases so every ds_load offset fits the 16-bit immediate field.
  const char* bg_base = &Bsm[lane * 32];
  const char* bx_base = &Bsm[65536 + lane * 32];

  v8f xacc[8], gacc[8];
#pragma unroll
  for (int j = 0; j < 8; ++j) { xacc[j] = (v8f)(0.f); gacc[j] = (v8f)(0.f); }

  // Double-buffered raw A tiles: k+1's HBM loads fly under k's cvt + 16 WMMAs.
  f32x4 graw[2][4], xraw[2][4];
  load_a_raw(agp + koff, graw[0]);
  load_a_raw(axp + koff, xraw[0]);

#pragma unroll
  for (int k = 0; k < 8; ++k) {
    const int cur = k & 1, nxt = cur ^ 1;
    if (k < 7) {
      load_a_raw(agp + (k + 1) * 32 + koff, graw[nxt]);
      load_a_raw(axp + (k + 1) * 32 + koff, xraw[nxt]);
    }
    v16bf ag = cvt_a_frag(graw[cur]);
    v16bf ax = cvt_a_frag(xraw[cur]);

    // Double-buffered B fragments: loads never overwrite regs a WMMA just read.
    v16bf bg = *(const v16bf*)(bg_base + k * 8192);
    v16bf bx = *(const v16bf*)(bx_base + k * 8192);
#pragma unroll
    for (int j = 0; j < 8; ++j) {
      v16bf bgn, bxn;
      if (j < 7) {
        bgn = *(const v16bf*)(bg_base + k * 8192 + (j + 1) * 1024);
        bxn = *(const v16bf*)(bx_base + k * 8192 + (j + 1) * 1024);
      }
      gacc[j] = __builtin_amdgcn_wmma_f32_16x16x32_bf16(
          false, ag, false, bg, (short)0, gacc[j], false, false);
      xacc[j] = __builtin_amdgcn_wmma_f32_16x16x32_bf16(
          false, ax, false, bx, (short)0, xacc[j], false, false);
      bg = bgn; bx = bxn;
    }
  }

  // ---- epilogue: fold biases, relu, psi projection, sigmoid, scale ----
  const float* vws  = (const float*)(ws + 131072);
  const int col0 = lane & 15;
  float bgl[8], bxl[8], wpl[8];
#pragma unroll
  for (int j = 0; j < 8; ++j) {
    int n = j * 16 + col0;
    bgl[j] = vws[n];
    bxl[j] = vws[128 + n];
    wpl[j] = vws[256 + n];
  }
  const float bpf = vws[384];

  float psi_part[8];
#pragma unroll
  for (int r = 0; r < 8; ++r) psi_part[r] = 0.f;

#pragma unroll
  for (int j = 0; j < 8; ++j) {
#pragma unroll
    for (int r = 0; r < 8; ++r) {
      float xv = xacc[j][r] + bxl[j];                      // x branch (post-BN)
      float sv = fmaxf(gacc[j][r] + bgl[j] + xv, 0.f);     // relu(g + x)
      xacc[j][r] = xv;
      psi_part[r] = fmaf(sv, wpl[j], psi_part[r]);
    }
  }

  // Butterfly-reduce psi over the 16 lanes of each half; C-layout rows match halves.
  float psi[8];
#pragma unroll
  for (int r = 0; r < 8; ++r) {
    float v = psi_part[r];
    v += __shfl_xor(v, 1, 32);
    v += __shfl_xor(v, 2, 32);
    v += __shfl_xor(v, 4, 32);
    v += __shfl_xor(v, 8, 32);
    psi[r] = 1.f / (1.f + __expf(-(v + bpf)));
  }

  float* obase = out + ((size_t)(rowbase + ((lane >> 4) << 3)) << 7) + col0;
#pragma unroll
  for (int r = 0; r < 8; ++r) {
    float pr = psi[r];
#pragma unroll
    for (int j = 0; j < 8; ++j)
      __builtin_nontemporal_store(xacc[j][r] * pr,
                                  obase + ((size_t)r << 7) + (j << 4));
  }
}

extern "C" void kernel_launch(void* const* d_in, const int* in_sizes, int n_in,
                              void* d_out, int out_size, void* d_ws, size_t ws_size,
                              hipStream_t stream)
{
  const float* input_g = (const float*)d_in[0];
  const float* input_x = (const float*)d_in[1];
  const float* Wg      = (const float*)d_in[2];
  const float* bg      = (const float*)d_in[3];
  const float* gamma_g = (const float*)d_in[4];
  const float* beta_g  = (const float*)d_in[5];
  const float* mean_g  = (const float*)d_in[6];
  const float* var_g   = (const float*)d_in[7];
  const float* Wx      = (const float*)d_in[8];
  const float* bx      = (const float*)d_in[9];
  const float* gamma_x = (const float*)d_in[10];
  const float* beta_x  = (const float*)d_in[11];
  const float* mean_x  = (const float*)d_in[12];
  const float* var_x   = (const float*)d_in[13];
  const float* Wp      = (const float*)d_in[14];
  const float* bp      = (const float*)d_in[15];
  const float* gamma_p = (const float*)d_in[16];
  const float* beta_p  = (const float*)d_in[17];
  const float* mean_p  = (const float*)d_in[18];
  const float* var_p   = (const float*)d_in[19];

  char* ws = (char*)d_ws;

  prep_weights<<<128, 256, 0, stream>>>(Wg, gamma_g, var_g, Wx, gamma_x, var_x,
                                        (uint32_t*)ws);
  prep_vectors<<<1, 128, 0, stream>>>(bg, gamma_g, beta_g, mean_g, var_g,
                                      bx, gamma_x, beta_x, mean_x, var_x,
                                      Wp, bp, gamma_p, beta_p, mean_p, var_p,
                                      (float*)(ws + 131072));
  attn_gate_main<<<NROW / 128, 256, 0, stream>>>(input_g, input_x, ws, (float*)d_out);
}
